// EpisodicMemory_5136780886052
// MI455X (gfx1250) — matrix operand order repaired and checked by reading, not compiled, not run on hardware
//
#include <hip/hip_runtime.h>

// ---------------- problem constants ----------------
#define H_    512
#define S_    64
#define B_    1024
#define SHID  120
#define FEAT_ 2048   // 4*H
#define G3H   1536   // 3*H
#define MROWS (B_ * S_)   // 65536

// ---------------- CDNA5 WMMA helpers ----------------
typedef __attribute__((ext_vector_type(16))) __bf16 bf16x16;
typedef __attribute__((ext_vector_type(8)))  __bf16 bf16x8;
typedef __attribute__((ext_vector_type(8)))  float  f32x8;

union BF16Frag { bf16x16 v; bf16x8 h[2]; };

__device__ __forceinline__ f32x8 wmma_bf16(bf16x16 a, bf16x16 b, f32x8 c) {
  // D(16x16,f32) = A(16x32,bf16) * B(32x16,bf16) + C
  return __builtin_amdgcn_wmma_f32_16x16x32_bf16(
      /*neg_a=*/false, a, /*neg_b=*/false, b,
      /*c_mod=*/(short)0, c, /*reuse_a=*/false, /*reuse_b=*/false);
}

// A fragment from row-major [16+][ld] bf16 tile in LDS.
// lane L<16 : row L, K {0..7, 16..23}; lane L>=16 : row L-16, K {8..15, 24..31}
__device__ __forceinline__ bf16x16 frag_a_lds(const __bf16* base, int ld) {
  const int lane = threadIdx.x & 31;
  const int lh   = lane >> 4;
  const __bf16* p = base + (lane & 15) * ld + lh * 8;
  BF16Frag u;
  u.h[0] = *(const bf16x8*)(p);
  u.h[1] = *(const bf16x8*)(p + 16);
  return u.v;
}

// B fragment: weights stored [N][K] (output-neuron rows, contiguous K).
// lane L<16 : col L, K {0..7, 16..23} ; lane L>=16 : col L-16, K {8..15, 24..31}
__device__ __forceinline__ bf16x16 frag_b_lds(const __bf16* base, int ld) {
  const int lane = threadIdx.x & 31;
  const int lh   = lane >> 4;
  const __bf16* p = base + (lane & 15) * ld + lh * 8;
  BF16Frag u;
  u.h[0] = *(const bf16x8*)(p);
  u.h[1] = *(const bf16x8*)(p + 16);
  return u.v;
}

__device__ __forceinline__ float sigmoidf_(float x) {
  return 1.f / (1.f + __expf(-x));
}

// ---------------- async global->LDS copy (CDNA5 ASYNCcnt path) ----------------
// Probe-verified: builtins exist; pointee type is an int vector
// (vector_size 16 for b128, 8 for b64), global src in AS1, LDS dst in AS3.
#if __has_builtin(__builtin_amdgcn_global_load_async_to_lds_b128)
#define HAVE_ASYNC_LDS 1
#else
#define HAVE_ASYNC_LDS 0
#endif

typedef int intx4 __attribute__((vector_size(4 * sizeof(int))));
typedef int intx2 __attribute__((vector_size(2 * sizeof(int))));
typedef __attribute__((address_space(1))) intx4* g4p;
typedef __attribute__((address_space(3))) intx4* l4p;
typedef __attribute__((address_space(1))) intx2* g2p;
typedef __attribute__((address_space(3))) intx2* l2p;

__device__ __forceinline__ void cp16_async(void* lds, const void* gsrc) {
#if HAVE_ASYNC_LDS
  __builtin_amdgcn_global_load_async_to_lds_b128((g4p)(gsrc), (l4p)(lds), 0, 0);
#else
  *(bf16x8*)lds = *(const bf16x8*)gsrc;
#endif
}

__device__ __forceinline__ void cp8_async(void* lds, const void* gsrc) {
#if HAVE_ASYNC_LDS && __has_builtin(__builtin_amdgcn_global_load_async_to_lds_b64)
  __builtin_amdgcn_global_load_async_to_lds_b64((g2p)(gsrc), (l2p)(lds), 0, 0);
#else
  *(unsigned long long*)lds = *(const unsigned long long*)gsrc;
#endif
}

// wait until ASYNCcnt <= n (plain SOPP; harmless when counter unused)
#define WAIT_ASYNC(n) asm volatile("s_wait_asynccnt " #n ::: "memory")

// ---------------- prep kernels ----------------
__global__ void cvt_f32_to_bf16(const float* __restrict__ s, __bf16* __restrict__ d, size_t n) {
  size_t i  = (size_t)blockIdx.x * blockDim.x + threadIdx.x;
  size_t st = (size_t)gridDim.x * blockDim.x;
  for (; i < n; i += st) d[i] = (__bf16)s[i];
}
__global__ void zero_bf16(__bf16* __restrict__ d, size_t n) {
  size_t i  = (size_t)blockIdx.x * blockDim.x + threadIdx.x;
  size_t st = (size_t)gridDim.x * blockDim.x;
  for (; i < n; i += st) d[i] = (__bf16)0.f;
}
__global__ void zero_f32(float* __restrict__ d, size_t n) {
  size_t i  = (size_t)blockIdx.x * blockDim.x + threadIdx.x;
  size_t st = (size_t)gridDim.x * blockDim.x;
  for (; i < n; i += st) d[i] = 0.f;
}

// ---------------- gate kernel: G = sigmoid(fc2(tanh(fc1(feat)))) ----------------
// One block per batch b: 4 waves x 16 rows = 64 rows (all s of this b).
// N padded to 128 (8 WMMA n-tiles), K = 2048 in chunks of 32.
__global__ __launch_bounds__(128)
void gate_kernel(const float* __restrict__ C, const float* __restrict__ Q,
                 const float* __restrict__ Mprev,
                 const __bf16* __restrict__ fc1_bf,  // [128][2048], rows 120..127 zero
                 const float*  __restrict__ fc1_b,   // [120]
                 const float*  __restrict__ fc2_w,   // [120]
                 const float*  __restrict__ fc2_b,   // [1]
                 float* __restrict__ Gout)           // [B*S]
{
  __shared__ float sQ[H_];
  __shared__ float sM[H_];
  __shared__ __align__(16) __bf16 sFeat[64][32];
  __shared__ __align__(16) __bf16 sW[128][32];

  const int b    = blockIdx.x;
  const int tid  = threadIdx.x;
  const int wave = tid >> 5;
  const int lane = tid & 31;
  const int lh   = lane >> 4;

  for (int i = tid; i < H_; i += 128) {
    sQ[i] = Q[(size_t)b * H_ + i];
    sM[i] = Mprev[(size_t)b * H_ + i];
  }
  __syncthreads();

  f32x8 acc[8];
#pragma unroll
  for (int t = 0; t < 8; ++t)
#pragma unroll
    for (int j = 0; j < 8; ++j) acc[t][j] = 0.f;

  const float* Crow = C + (size_t)b * S_ * H_;

#pragma unroll 2
  for (int k0 = 0; k0 < FEAT_; k0 += 32) {
    const int q  = k0 >> 9;     // feature quadrant: CQ, CM, |C-Q|, |C-M|
    const int hb = k0 & (H_ - 1);

    // stage fc1 weight chunk (128 rows x 32) via async global->LDS
    {
      const __bf16* wp = fc1_bf + (size_t)tid * FEAT_ + k0;
      cp16_async(&sW[tid][0],  wp);
      cp16_async(&sW[tid][8],  wp + 8);
      cp16_async(&sW[tid][16], wp + 16);
      cp16_async(&sW[tid][24], wp + 24);
    }
    // stage feat chunk (64 rows x 32) built on the fly (VALU -> ds_store)
    {
      const int r  = tid >> 1;
      const int kk = (tid & 1) * 16;
      const float* cp = Crow + (size_t)r * H_ + hb + kk;
      const float* up = ((q & 1) == 0) ? (sQ + hb + kk) : (sM + hb + kk);
#pragma unroll
      for (int e = 0; e < 16; ++e) {
        const float c = cp[e], u = up[e];
        const float v = (q < 2) ? (c * u) : fabsf(c - u);
        sFeat[r][kk + e] = (__bf16)v;
      }
      if (k0 + 32 < FEAT_) __builtin_prefetch(cp + 32, 0, 1);
    }
    WAIT_ASYNC(0x0);
    __syncthreads();

    // all B fragments first, then back-to-back WMMAs (two groups of 4)
    const bf16x16 a = frag_a_lds(&sFeat[wave * 16][0], 32);
    bf16x16 bfr[4];
#pragma unroll
    for (int t = 0; t < 4; ++t) bfr[t] = frag_b_lds(&sW[t * 16][0], 32);
#pragma unroll
    for (int t = 0; t < 4; ++t) acc[t] = wmma_bf16(a, bfr[t], acc[t]);
#pragma unroll
    for (int t = 0; t < 4; ++t) bfr[t] = frag_b_lds(&sW[(t + 4) * 16][0], 32);
#pragma unroll
    for (int t = 0; t < 4; ++t) acc[t + 4] = wmma_bf16(a, bfr[t], acc[t + 4]);
    __syncthreads();
  }

  // epilogue: tanh -> dot with fc2 -> 16-lane reduce -> sigmoid
  float fc1bv[8], fc2v[8];
#pragma unroll
  for (int t = 0; t < 8; ++t) {
    const int col = t * 16 + (lane & 15);
    fc1bv[t] = (col < SHID) ? fc1_b[col] : 0.f;
    fc2v[t]  = (col < SHID) ? fc2_w[col] : 0.f;
  }
  const float b2 = fc2_b[0];

#pragma unroll
  for (int r = 0; r < 8; ++r) {
    float p = 0.f;
#pragma unroll
    for (int t = 0; t < 8; ++t) p += tanhf(acc[t][r] + fc1bv[t]) * fc2v[t];
#pragma unroll
    for (int m = 1; m <= 8; m <<= 1) p += __shfl_xor(p, m, 32);
    if ((lane & 15) == 0) {
      const int s = wave * 16 + r + 8 * lh;
      Gout[(size_t)b * S_ + s] = sigmoidf_(p + b2);
    }
  }
}

// ---------------- xg kernel: xg_all = C @ W_ih^T + b_ih ----------------
// M=65536, N=1536, K=512. Block tile 64x128, 8 waves (4 m-subtiles x 2 n-halves).
// Double-buffered async global->LDS staging (3 async b128 per thread per chunk).
__global__ __launch_bounds__(256)
void xg_kernel(const __bf16* __restrict__ Cbf,   // [65536][512]
               const __bf16* __restrict__ Wih,   // [1536][512]
               const float*  __restrict__ b_ih,  // [1536]
               float* __restrict__ xg)           // [65536][1536]
{
  __shared__ __align__(16) __bf16 sA[2][64][32];
  __shared__ __align__(16) __bf16 sB[2][128][32];

  const int tid  = threadIdx.x;
  const int lane = tid & 31;
  const int wave = tid >> 5;
  const int wm   = wave & 3;
  const int wn   = wave >> 2;
  const int lh   = lane >> 4;
  const int mb   = blockIdx.x * 64;
  const int nb   = blockIdx.y * 128;

  f32x8 acc[4];
#pragma unroll
  for (int t = 0; t < 4; ++t)
#pragma unroll
    for (int j = 0; j < 8; ++j) acc[t][j] = 0.f;

  const int ra = tid >> 2, ka = (tid & 3) * 8;   // A-stage coords (1 x b128)
  const int rb = tid >> 1, kb = (tid & 1) * 16;  // B-stage coords (2 x b128)

  auto issue = [&](int buf, int k0) {
    cp16_async(&sA[buf][ra][ka], Cbf + (size_t)(mb + ra) * H_ + k0 + ka);
    const __bf16* p = Wih + (size_t)(nb + rb) * H_ + k0 + kb;
    cp16_async(&sB[buf][rb][kb], p);
    cp16_async(&sB[buf][rb][kb + 8], p + 8);
  };

  issue(0, 0);
#pragma unroll 2
  for (int i = 0; i < H_ / 32; ++i) {
    if (i + 1 < H_ / 32) {
      issue((i + 1) & 1, (i + 1) * 32);
      WAIT_ASYNC(0x3);            // own chunk-i copies done; next 3 in flight
    } else {
      WAIT_ASYNC(0x0);
    }
    __syncthreads();              // all waves' chunk-i copies visible

    const bf16x16 a = frag_a_lds(&sA[i & 1][wm * 16][0], 32);
    bf16x16 bfr[4];
#pragma unroll
    for (int t = 0; t < 4; ++t)
      bfr[t] = frag_b_lds(&sB[i & 1][wn * 64 + t * 16][0], 32);
#pragma unroll
    for (int t = 0; t < 4; ++t) acc[t] = wmma_bf16(a, bfr[t], acc[t]);
    __syncthreads();              // all waves done reading -> safe to overwrite
  }

#pragma unroll
  for (int t = 0; t < 4; ++t) {
    const int col  = nb + wn * 64 + t * 16 + (lane & 15);
    const float bv = b_ih[col];
#pragma unroll
    for (int r = 0; r < 8; ++r) {
      const int m = mb + wm * 16 + r + 8 * lh;
      xg[(size_t)m * G3H + col] = acc[t][r] + bv;
    }
  }
}

// ---------------- GRU step kernel (launched 64x, sequential) ----------------
// Per step: hg = h @ W_hh^T (+b_hh fused), then full gating.
// Block tile: 64 batch-rows x 32 gate-cols; r/z/n blocks computed together so
// the n-gate (needs hn separate from xn) fuses in the epilogue.
// Double-buffered async staging: 1 x b128 (A) + 3 x b64 (B) per thread per chunk.
__global__ __launch_bounds__(256)
void gru_step_kernel(const __bf16* __restrict__ hbf_in,  // [1024][512]
                     const float*  __restrict__ hf_in,   // [1024][512]
                     const __bf16* __restrict__ Whh,     // [1536][512]
                     const float*  __restrict__ b_hh,    // [1536]
                     const float*  __restrict__ xg,      // [65536][1536]
                     const float*  __restrict__ Gg,      // [1024][64]
                     int t,
                     float*  __restrict__ hf_out,        // [1024][512]
                     __bf16* __restrict__ hbf_out)       // [1024][512]
{
  __shared__ __align__(16) __bf16 sA[2][64][32];
  __shared__ __align__(16) __bf16 sB[2][96][32];  // rows: [0,32)=r, [32,64)=z, [64,96)=n

  const int tid  = threadIdx.x;
  const int lane = tid & 31;
  const int wave = tid >> 5;
  const int wm   = wave & 3;
  const int wg   = wave >> 2;
  const int lh   = lane >> 4;
  const int mb   = blockIdx.x * 64;
  const int nc   = blockIdx.y * 32;

  f32x8 accr, accz, accn;
#pragma unroll
  for (int j = 0; j < 8; ++j) { accr[j] = 0.f; accz[j] = 0.f; accn[j] = 0.f; }

  const int ra = tid >> 2, ka = (tid & 3) * 8;   // A-stage coords

  auto issue = [&](int buf, int k0) {
    cp16_async(&sA[buf][ra][ka], hbf_in + (size_t)(mb + ra) * H_ + k0 + ka);
#pragma unroll
    for (int j = 0; j < 3; ++j) {                // B: 768 x 8B units over 256 threads
      const int u  = tid * 3 + j;
      const int r  = u >> 3;                     // LDS row 0..95
      const int co = (u & 7) * 4;                // element offset within row
      const int nr = nc + (r & 31) + (r >> 5) * H_;
      cp8_async(&sB[buf][r][co], Whh + (size_t)nr * H_ + k0 + co);
    }
  };

  issue(0, 0);
#pragma unroll 2
  for (int i = 0; i < H_ / 32; ++i) {
    if (i + 1 < H_ / 32) {
      issue((i + 1) & 1, (i + 1) * 32);
      WAIT_ASYNC(0x4);            // 4 ops per thread per chunk
    } else {
      WAIT_ASYNC(0x0);
    }
    __syncthreads();

    const bf16x16 a = frag_a_lds(&sA[i & 1][wm * 16][0], 32);
    bf16x16 br_ = frag_b_lds(&sB[i & 1][ 0 + wg * 16][0], 32);
    bf16x16 bz_ = frag_b_lds(&sB[i & 1][32 + wg * 16][0], 32);
    bf16x16 bn_ = frag_b_lds(&sB[i & 1][64 + wg * 16][0], 32);
    accr = wmma_bf16(a, br_, accr);
    accz = wmma_bf16(a, bz_, accz);
    accn = wmma_bf16(a, bn_, accn);
    __syncthreads();
  }

  const int c  = nc + wg * 16 + (lane & 15);
  const float br = b_hh[c], bz = b_hh[H_ + c], bn = b_hh[2 * H_ + c];

#pragma unroll
  for (int r = 0; r < 8; ++r) {
    const int m = mb + wm * 16 + r + 8 * lh;
    const size_t xb = ((size_t)m * S_ + t) * G3H + c;
    const float xr = xg[xb], xz = xg[xb + H_], xn = xg[xb + 2 * H_];
    const float hr = accr[r] + br;
    const float hz = accz[r] + bz;
    const float hn = accn[r] + bn;
    const float rg = sigmoidf_(xr + hr);
    const float zg = sigmoidf_(xz + hz);
    const float nn = tanhf(xn + rg * hn);
    const float hp = hf_in[(size_t)m * H_ + c];
    const float hgru = (1.f - zg) * nn + zg * hp;
    const float g  = Gg[(size_t)m * S_ + t];
    const float hnew = g * hgru + (1.f - g) * hp;
    hf_out[(size_t)m * H_ + c]  = hnew;
    hbf_out[(size_t)m * H_ + c] = (__bf16)hnew;
  }
}

// ---------------- workspace layout (bytes, all 256-aligned) ----------------
static const size_t OFF_XG  = 0;                                     // 402653184
static const size_t OFF_CBF = OFF_XG  + (size_t)MROWS * G3H * 4;     // + 67108864
static const size_t OFF_WIH = OFF_CBF + (size_t)MROWS * H_ * 2;      // + 1572864
static const size_t OFF_WHH = OFF_WIH + (size_t)G3H * H_ * 2;        // + 1572864
static const size_t OFF_FC1 = OFF_WHH + (size_t)G3H * H_ * 2;        // + 524288
static const size_t OFF_G   = OFF_FC1 + (size_t)128 * FEAT_ * 2;     // + 262144
static const size_t OFF_HF  = OFF_G   + (size_t)MROWS * 4;           // + 2*2097152
static const size_t OFF_HB  = OFF_HF  + (size_t)2 * B_ * H_ * 4;     // + 2*1048576

extern "C" void kernel_launch(void* const* d_in, const int* in_sizes, int n_in,
                              void* d_out, int out_size, void* d_ws, size_t ws_size,
                              hipStream_t stream) {
  const float* C     = (const float*)d_in[0];
  const float* Q     = (const float*)d_in[1];
  const float* Mprev = (const float*)d_in[2];
  const float* fc1_w = (const float*)d_in[3];
  const float* fc1_b = (const float*)d_in[4];
  const float* fc2_w = (const float*)d_in[5];
  const float* fc2_b = (const float*)d_in[6];
  const float* W_ih  = (const float*)d_in[7];
  const float* W_hh  = (const float*)d_in[8];
  const float* b_ih  = (const float*)d_in[9];
  const float* b_hh  = (const float*)d_in[10];

  char* ws = (char*)d_ws;
  float*  xg     = (float*)(ws + OFF_XG);
  __bf16* Cbf    = (__bf16*)(ws + OFF_CBF);
  __bf16* WihBf  = (__bf16*)(ws + OFF_WIH);
  __bf16* WhhBf  = (__bf16*)(ws + OFF_WHH);
  __bf16* fc1Bf  = (__bf16*)(ws + OFF_FC1);
  float*  Gw     = (float*)(ws + OFF_G);
  float*  hF[2]  = { (float*)(ws + OFF_HF), (float*)(ws + OFF_HF) + (size_t)B_ * H_ };
  __bf16* hB[2]  = { (__bf16*)(ws + OFF_HB), (__bf16*)(ws + OFF_HB) + (size_t)B_ * H_ };

  // ---- prep: bf16 conversions + zero initial hidden state ----
  cvt_f32_to_bf16<<<2048, 256, 0, stream>>>(C, Cbf, (size_t)MROWS * H_);
  cvt_f32_to_bf16<<<256, 256, 0, stream>>>(W_ih, WihBf, (size_t)G3H * H_);
  cvt_f32_to_bf16<<<256, 256, 0, stream>>>(W_hh, WhhBf, (size_t)G3H * H_);
  zero_bf16<<<64, 256, 0, stream>>>(fc1Bf, (size_t)128 * FEAT_);          // pad rows 120..127 = 0
  cvt_f32_to_bf16<<<64, 256, 0, stream>>>(fc1_w, fc1Bf, (size_t)SHID * FEAT_);
  zero_f32<<<64, 256, 0, stream>>>(hF[0], (size_t)B_ * H_);
  zero_bf16<<<64, 256, 0, stream>>>(hB[0], (size_t)B_ * H_);

  // ---- gate scores G(B,S) ----
  gate_kernel<<<B_, 128, 0, stream>>>(C, Q, Mprev, fc1Bf, fc1_b, fc2_w, fc2_b, Gw);

  // ---- xg_all = C @ W_ih^T + b_ih (one big WMMA GEMM) ----
  xg_kernel<<<dim3(MROWS / 64, G3H / 128), 256, 0, stream>>>(Cbf, WihBf, b_ih, xg);

  // ---- sequential GRU scan: 64 dependent steps ----
  for (int t = 0; t < 64; ++t) {
    const int pi = t & 1, po = (t + 1) & 1;
    float* hout = (t == 63) ? (float*)d_out : hF[po];
    gru_step_kernel<<<dim3(B_ / 64, H_ / 32), 256, 0, stream>>>(
        hB[pi], hF[pi], WhhBf, b_hh, xg, Gw, t, hout, hB[po]);
  }
}
